// MultiHeadMoeBlock_32796370272626
// MI455X (gfx1250) — matrix operand order repaired and checked by reading, not compiled, not run on hardware
//
#include <hip/hip_runtime.h>

// ---------------------------------------------------------------------------
// MultiHead-MoE block for gfx1250 (MI455X), bf16 WMMA (v_wmma_f32_16x16x32_bf16)
// with f32 accumulation. All matrix math runs through WMMA; wave32 layouts per
// CDNA5 ISA 7.12.2.
// ---------------------------------------------------------------------------

typedef __attribute__((ext_vector_type(16))) __bf16 v16bf;
typedef __attribute__((ext_vector_type(8)))  __bf16 v8bf;
typedef __attribute__((ext_vector_type(8)))  float  v8f;

constexpr int kB  = 4;
constexpr int kL  = 1024;
constexpr int kH  = 1024;
constexpr int kE  = 16;
constexpr int kNH = 16;
constexpr int kD  = 64;                 // head dim == expert width
constexpr int kM  = kB * kL;            // 4096 rows for the two big GEMMs
constexpr int kNT = kM * kNH;           // 65536 sub-tokens

// ---- WMMA helpers ----------------------------------------------------------

__device__ __forceinline__ v8f wmma_bf16(v16bf a, v16bf b, v8f c) {
  // (neg_a, A, neg_b, B, c_mod, C, reuse_a, reuse_b)
  return __builtin_amdgcn_wmma_f32_16x16x32_bf16(false, a, false, b,
                                                 (short)0, c, false, false);
}

// A-fragment (16x32 bf16, M x K). Lanes 0-15 hold row M=lane with
// K {0..15}; lanes 16-31 hold K {16..31}. Two 16B chunks/lane.
__device__ __forceinline__ v16bf load_frag_a(const __bf16* base, int ldk,
                                             int row, int k0, int hf) {
  const __bf16* p = base + (size_t)row * ldk + k0 + hf * 16;
  v8bf c0 = *(const v8bf*)p;
  v8bf c1 = *(const v8bf*)(p + 8);
  return __builtin_shufflevector(c0, c1, 0, 1, 2, 3, 4, 5, 6, 7,
                                 8, 9, 10, 11, 12, 13, 14, 15);
}

// B-fragment (32x16 bf16, K x N) from an N-major (transposed) matrix Bt[N][K].
// Lanes 0-15 hold K=k0..k0+15 of column N=lane; lanes 16-31 hold K=k0+16..31.
__device__ __forceinline__ v16bf load_frag_b(const __bf16* bt, int ldk,
                                             int col, int k0, int hf) {
  return *(const v16bf*)(bt + (size_t)col * ldk + k0 + hf * 16);
}

// ---- prep kernels ----------------------------------------------------------

__global__ __launch_bounds__(256) void f32_to_bf16_kernel(
    const float* __restrict__ in, __bf16* __restrict__ out, int n) {
  int i4 = (blockIdx.x * 256 + threadIdx.x) * 4;
  if (i4 + 3 < n) {
    float4 v = *(const float4*)(in + i4);
    out[i4 + 0] = (__bf16)v.x;
    out[i4 + 1] = (__bf16)v.y;
    out[i4 + 2] = (__bf16)v.z;
    out[i4 + 3] = (__bf16)v.w;
  } else {
    for (int i = i4; i < n; ++i) out[i] = (__bf16)in[i];
  }
}

// W[R][C] f32 -> Wt[C][R] bf16 (batched over blockIdx.z). block = (32,8).
__global__ __launch_bounds__(256) void transpose_f32_to_bf16(
    const float* __restrict__ W, __bf16* __restrict__ Wt, int R, int C) {
  __shared__ float tile[32][33];
  const size_t bo = (size_t)blockIdx.z * R * C;
  const int c0 = blockIdx.x * 32, r0 = blockIdx.y * 32;
  const int tx = threadIdx.x, ty = threadIdx.y;
  for (int i = ty; i < 32; i += 8)
    tile[i][tx] = W[bo + (size_t)(r0 + i) * C + (c0 + tx)];
  __syncthreads();
  for (int i = ty; i < 32; i += 8)
    Wt[bo + (size_t)(c0 + i) * R + (r0 + tx)] = (__bf16)tile[tx][i];
}

// ---- big GEMM: C[M,N] = A_bf16[M,K] @ Bt_bf16[N,K]^T + bias ---------------
// 8 waves/block, block tile 64(M) x 128(N); wave tile 32x32 (2x2 WMMA accum).

__global__ __launch_bounds__(256) void gemm_bf16_kernel(
    const __bf16* __restrict__ A, const __bf16* __restrict__ Bt,
    const float* __restrict__ bias, float* __restrict__ Cf,
    __bf16* __restrict__ Cb, int M, int N, int K) {
  const int lane = threadIdx.x & 31;
  const int wv   = threadIdx.x >> 5;
  const int hf   = lane >> 4;
  const int l15  = lane & 15;
  const int mBase = blockIdx.x * 64 + (wv >> 2) * 32;
  const int nBase = blockIdx.y * 128 + (wv & 3) * 32;

  v8f acc[2][2] = {};
  for (int k0 = 0; k0 < K; k0 += 32) {
    // hint the A stream forward (emits global_prefetch)
    __builtin_prefetch(A + (size_t)(mBase + l15) * K + k0 + 256, 0, 1);
    v16bf a0 = load_frag_a(A, K, mBase + l15, k0, hf);
    v16bf a1 = load_frag_a(A, K, mBase + 16 + l15, k0, hf);
    v16bf b0 = load_frag_b(Bt, K, nBase + l15, k0, hf);
    v16bf b1 = load_frag_b(Bt, K, nBase + 16 + l15, k0, hf);
    acc[0][0] = wmma_bf16(a0, b0, acc[0][0]);
    acc[0][1] = wmma_bf16(a0, b1, acc[0][1]);
    acc[1][0] = wmma_bf16(a1, b0, acc[1][0]);
    acc[1][1] = wmma_bf16(a1, b1, acc[1][1]);
  }
  // C/D layout: lanes 0-15 -> (M=r, N=lane); lanes 16-31 -> (M=r+8, N=lane-16)
  for (int tm = 0; tm < 2; ++tm) {
    for (int tn = 0; tn < 2; ++tn) {
      const int col = nBase + tn * 16 + l15;
      const float bv = bias ? bias[col] : 0.0f;
#pragma unroll
      for (int r = 0; r < 8; ++r) {
        const int row = mBase + tm * 16 + hf * 8 + r;
        const float v = acc[tm][tn][r] + bv;
        if (Cf) Cf[(size_t)row * N + col] = v;
        if (Cb) Cb[(size_t)row * N + col] = (__bf16)v;
      }
    }
  }
}

// ---- router: softmax over 16 experts + top-2 -------------------------------

__global__ __launch_bounds__(256) void router_kernel(
    const float* __restrict__ h, const float* __restrict__ emb,
    int2* __restrict__ sel, float2* __restrict__ rw, int nTok) {
  __shared__ float se[kD * kE];
  for (int i = threadIdx.x; i < kD * kE; i += 256) se[i] = emb[i];
  __syncthreads();
  const int t = blockIdx.x * 256 + threadIdx.x;
  if (t >= nTok) return;
  float lg[kE];
#pragma unroll
  for (int e = 0; e < kE; ++e) lg[e] = 0.0f;
  const float* hr = h + (size_t)t * kD;
  for (int d = 0; d < kD; ++d) {
    const float hv = hr[d];
    const float* er = &se[d * kE];
#pragma unroll
    for (int e = 0; e < kE; ++e) lg[e] = fmaf(hv, er[e], lg[e]);
  }
  float mx = lg[0];
#pragma unroll
  for (int e = 1; e < kE; ++e) mx = fmaxf(mx, lg[e]);
  float s = 0.0f;
#pragma unroll
  for (int e = 0; e < kE; ++e) { lg[e] = __expf(lg[e] - mx); s += lg[e]; }
  const float inv = 1.0f / s;
  int i0 = 0, i1 = 0;
  float p0 = -1.0f, p1 = -1.0f;
#pragma unroll
  for (int e = 0; e < kE; ++e) {
    const float p = lg[e] * inv;
    if (p > p0) { p1 = p0; i1 = i0; p0 = p; i0 = e; }
    else if (p > p1) { p1 = p; i1 = e; }
  }
  sel[t] = make_int2(i0, i1);
  rw[t]  = make_float2(p0, p1);
}

// ---- fused expert MLP + top-2 combine --------------------------------------
// 128 tokens / block (4 waves x 32 tokens). Per expert: stage 3x(64x64) bf16
// transposed weights in LDS, run 3 WMMA layers with LDS transpose buffers
// between layers (C-layout -> A-layout), gate-weighted accumulate into comb.

__global__ __launch_bounds__(128) void expert_moe_kernel(
    const __bf16* __restrict__ hB, const int2* __restrict__ sel,
    const float2* __restrict__ rw,
    const __bf16* __restrict__ w1t, const __bf16* __restrict__ w2t,
    const __bf16* __restrict__ w3t,
    const float* __restrict__ b1, const float* __restrict__ b2,
    const float* __restrict__ b3,
    __bf16* __restrict__ combB) {
  __shared__ __bf16 wlds[3 * kD * kD];   // 24 KB: [w1t | w2t | w3t] for expert e
  __shared__ float  gates[kE * 128];     //  8 KB: gate[e][localTok]
  __shared__ __bf16 trb[4 * 32 * 72];    // 18 KB: per-wave 32x64 (+8 pad) tile

  const int tid  = threadIdx.x;
  const int lane = tid & 31;
  const int wv   = tid >> 5;    // 0..3
  const int hf   = lane >> 4;
  const int l15  = lane & 15;
  const int tokBase = blockIdx.x * 128;

  { // routing gates, one token per thread
    const int2   s = sel[tokBase + tid];
    const float2 w = rw[tokBase + tid];
#pragma unroll
    for (int e = 0; e < kE; ++e)
      gates[e * 128 + tid] = (s.x == e) ? w.x : ((s.y == e) ? w.y : 0.0f);
  }

  const int rowBase = tokBase + wv * 32;

  // persistent A-fragments of the h tile (32 rows x K=64) — reused for all 16 experts
  v16bf aH[2][2];
#pragma unroll
  for (int m = 0; m < 2; ++m)
#pragma unroll
    for (int kk = 0; kk < 2; ++kk)
      aH[m][kk] = load_frag_a(hB, kD, rowBase + m * 16 + l15, kk * 32, hf);

  v8f comb[2][4] = {};
  __bf16* myTr = trb + wv * (32 * 72);

  for (int e = 0; e < kE; ++e) {
    __syncthreads();   // everyone done reading wlds from previous expert
    { // stage 3 x 4096 bf16 = 1536 16B-chunks, 12 per thread
      const v8bf* s1 = (const v8bf*)(w1t + (size_t)e * kD * kD);
      const v8bf* s2 = (const v8bf*)(w2t + (size_t)e * kD * kD);
      const v8bf* s3 = (const v8bf*)(w3t + (size_t)e * kD * kD);
      v8bf* d0 = (v8bf*)wlds;
      for (int i = tid; i < 512; i += 128) {
        d0[i] = s1[i]; d0[512 + i] = s2[i]; d0[1024 + i] = s3[i];
      }
    }
    __syncthreads();

    // ---- layer 1: relu(h @ w1 + b1) -> trb
#pragma unroll
    for (int n = 0; n < 4; ++n) {
      v16bf bf0 = load_frag_b(wlds, kD, n * 16 + l15, 0, hf);
      v16bf bf1 = load_frag_b(wlds, kD, n * 16 + l15, 32, hf);
      const float bb = b1[e * kD + n * 16 + l15];
#pragma unroll
      for (int m = 0; m < 2; ++m) {
        v8f t = {0.f, 0.f, 0.f, 0.f, 0.f, 0.f, 0.f, 0.f};
        t = wmma_bf16(aH[m][0], bf0, t);
        t = wmma_bf16(aH[m][1], bf1, t);
#pragma unroll
        for (int r = 0; r < 8; ++r) {
          float v = t[r] + bb;
          v = v > 0.0f ? v : 0.0f;
          myTr[(m * 16 + hf * 8 + r) * 72 + n * 16 + l15] = (__bf16)v;
        }
      }
    }
    v16bf aT[2][2];
#pragma unroll
    for (int m = 0; m < 2; ++m)
#pragma unroll
      for (int kk = 0; kk < 2; ++kk)
        aT[m][kk] = load_frag_a(myTr, 72, m * 16 + l15, kk * 32, hf);

    // ---- layer 2: relu(t1 @ w2 + b2) -> trb
#pragma unroll
    for (int n = 0; n < 4; ++n) {
      v16bf bf0 = load_frag_b(wlds + kD * kD, kD, n * 16 + l15, 0, hf);
      v16bf bf1 = load_frag_b(wlds + kD * kD, kD, n * 16 + l15, 32, hf);
      const float bb = b2[e * kD + n * 16 + l15];
#pragma unroll
      for (int m = 0; m < 2; ++m) {
        v8f t = {0.f, 0.f, 0.f, 0.f, 0.f, 0.f, 0.f, 0.f};
        t = wmma_bf16(aT[m][0], bf0, t);
        t = wmma_bf16(aT[m][1], bf1, t);
#pragma unroll
        for (int r = 0; r < 8; ++r) {
          float v = t[r] + bb;
          v = v > 0.0f ? v : 0.0f;
          myTr[(m * 16 + hf * 8 + r) * 72 + n * 16 + l15] = (__bf16)v;
        }
      }
    }
    v16bf aU[2][2];
#pragma unroll
    for (int m = 0; m < 2; ++m)
#pragma unroll
      for (int kk = 0; kk < 2; ++kk)
        aU[m][kk] = load_frag_a(myTr, 72, m * 16 + l15, kk * 32, hf);

    // ---- layer 3: (t2 @ w3 + b3), gate-weighted accumulate
#pragma unroll
    for (int n = 0; n < 4; ++n) {
      v16bf bf0 = load_frag_b(wlds + 2 * kD * kD, kD, n * 16 + l15, 0, hf);
      v16bf bf1 = load_frag_b(wlds + 2 * kD * kD, kD, n * 16 + l15, 32, hf);
      const float bb = b3[e * kD + n * 16 + l15];
#pragma unroll
      for (int m = 0; m < 2; ++m) {
        v8f t = {0.f, 0.f, 0.f, 0.f, 0.f, 0.f, 0.f, 0.f};
        t = wmma_bf16(aU[m][0], bf0, t);
        t = wmma_bf16(aU[m][1], bf1, t);
#pragma unroll
        for (int r = 0; r < 8; ++r) {
          const int rl = wv * 32 + m * 16 + hf * 8 + r;
          const float g = gates[e * 128 + rl];
          comb[m][n][r] += g * (t[r] + bb);
        }
      }
    }
  }

  // comb [kNT, 64] flat == [4096, 1024] row-major -> feed GEMM2 directly
#pragma unroll
  for (int m = 0; m < 2; ++m)
#pragma unroll
    for (int n = 0; n < 4; ++n)
#pragma unroll
      for (int r = 0; r < 8; ++r)
        combB[(size_t)(rowBase + m * 16 + hf * 8 + r) * kD + n * 16 + l15] =
            (__bf16)comb[m][n][r];
}

// ---------------------------------------------------------------------------

extern "C" void kernel_launch(void* const* d_in, const int* in_sizes, int n_in,
                              void* d_out, int out_size, void* d_ws,
                              size_t ws_size, hipStream_t stream) {
  (void)in_sizes; (void)n_in; (void)out_size; (void)ws_size;
  const float* x   = (const float*)d_in[0];
  const float* Wmh = (const float*)d_in[1];
  const float* bmh = (const float*)d_in[2];
  const float* emb = (const float*)d_in[3];
  const float* w1  = (const float*)d_in[4];
  const float* b1  = (const float*)d_in[5];
  const float* w2  = (const float*)d_in[6];
  const float* b2  = (const float*)d_in[7];
  const float* w3  = (const float*)d_in[8];
  const float* b3  = (const float*)d_in[9];
  const float* Wm  = (const float*)d_in[10];
  const float* bm  = (const float*)d_in[11];
  float* out = (float*)d_out;

  // workspace layout (~46 MB)
  char* ws = (char*)d_ws;
  float*  h_f32 = (float*)(ws);                                  // 16 MB
  __bf16* h_bf  = (__bf16*)(ws + (size_t)16 * 1024 * 1024);      //  8 MB
  __bf16* x_bf  = (__bf16*)(ws + (size_t)24 * 1024 * 1024);      //  8 MB
  __bf16* WmhT  = (__bf16*)(ws + (size_t)32 * 1024 * 1024);      //  2 MB
  __bf16* WmT   = (__bf16*)(ws + (size_t)34 * 1024 * 1024);      //  2 MB
  __bf16* combB = (__bf16*)(ws + (size_t)36 * 1024 * 1024);      //  8 MB
  __bf16* w1t   = (__bf16*)(ws + (size_t)44 * 1024 * 1024);      // 128 KB
  __bf16* w2t   = (__bf16*)(ws + (size_t)44 * 1024 * 1024 + 131072);
  __bf16* w3t   = (__bf16*)(ws + (size_t)44 * 1024 * 1024 + 262144);
  int2*   sel   = (int2*)(ws + (size_t)45 * 1024 * 1024);        // 512 KB
  float2* rw    = (float2*)(ws + (size_t)45 * 1024 * 1024 + 524288);

  // 1) x -> bf16
  {
    const int n = kM * kH;
    f32_to_bf16_kernel<<<n / (256 * 4), 256, 0, stream>>>(x, x_bf, n);
  }
  // 2) transpose+convert big weights: W[K][N] -> Wt[N][K] bf16
  {
    dim3 blk(32, 8);
    transpose_f32_to_bf16<<<dim3(kH / 32, kH / 32, 1), blk, 0, stream>>>(
        Wmh, WmhT, kH, kH);
    transpose_f32_to_bf16<<<dim3(kH / 32, kH / 32, 1), blk, 0, stream>>>(
        Wm, WmT, kH, kH);
    // expert weights: w[e][k][n] -> wt[e][n][k]
    transpose_f32_to_bf16<<<dim3(2, 2, kE), blk, 0, stream>>>(w1, w1t, kD, kD);
    transpose_f32_to_bf16<<<dim3(2, 2, kE), blk, 0, stream>>>(w2, w2t, kD, kD);
    transpose_f32_to_bf16<<<dim3(2, 2, kE), blk, 0, stream>>>(w3, w3t, kD, kD);
  }
  // 3) GEMM1: h = x @ Wmh + bmh  (writes f32 for router + bf16 for experts)
  gemm_bf16_kernel<<<dim3(kM / 64, kH / 128), 256, 0, stream>>>(
      x_bf, WmhT, bmh, h_f32, h_bf, kM, kH, kH);
  // 4) router top-2
  router_kernel<<<kNT / 256, 256, 0, stream>>>(h_f32, emb, sel, rw, kNT);
  // 5) fused expert MLPs + combine -> comb (bf16, == [4096,1024] row-major)
  expert_moe_kernel<<<kNT / 128, 128, 0, stream>>>(
      h_bf, sel, rw, w1t, w2t, w3t, b1, b2, b3, combB);
  // 6) GEMM2: out = comb @ Wm + bm  (f32 output)
  gemm_bf16_kernel<<<dim3(kM / 64, kH / 128), 256, 0, stream>>>(
      combB, WmT, bm, out, nullptr, kM, kH, kH);
}
